// QThreeLayer_BNFold_79147657331048
// MI455X (gfx1250) — 8-bit matrix operand order repaired and checked by reading, compiled, not run on hardware
//
#include <hip/hip_runtime.h>
#include <math.h>

typedef __attribute__((ext_vector_type(8))) int v8i;

#define TM 128
#define TN 128
#define LDT 80

__device__ __forceinline__ void async_ld16(unsigned lds, const void* g) {
    asm volatile("global_load_async_to_lds_b128 %0, %1, off"
                 :: "v"(lds), "v"(g) : "memory");
}
__device__ __forceinline__ void wait_async0() {
    asm volatile("s_wait_asynccnt 0x0" ::: "memory");
}

__global__ void k_init(float* p) {
    if (threadIdx.x < 16) p[threadIdx.x] = 0.0f;
}

__global__ __launch_bounds__(256) void k_prep(const float* __restrict__ w,
                                              const float* __restrict__ b,
                                              const float* __restrict__ g,
                                              const float* __restrict__ be,
                                              const float* __restrict__ m,
                                              const float* __restrict__ v,
                                              int K, float* __restrict__ bf,
                                              float* __restrict__ amax_w_slot) {
    const int n = blockIdx.x;
    const float inv = g[n] / sqrtf(v[n] + 1e-5f);
    if (threadIdx.x == 0) bf[n] = (b[n] - m[n]) * inv + be[n];
    const float* wr = w + (size_t)n * K;
    float mx = 0.0f;
    for (int k = threadIdx.x; k < K; k += 256) mx = fmaxf(mx, fabsf(wr[k] * inv));
    __shared__ float red[256];
    red[threadIdx.x] = mx;
    __syncthreads();
    for (int s = 128; s > 0; s >>= 1) {
        if (threadIdx.x < s) red[threadIdx.x] = fmaxf(red[threadIdx.x], red[threadIdx.x + s]);
        __syncthreads();
    }
    if (threadIdx.x == 0)
        atomicMax((unsigned int*)amax_w_slot, __float_as_uint(red[0]));
}

__global__ __launch_bounds__(256) void k_quant_w(const float* __restrict__ w,
                                                 const float* __restrict__ g,
                                                 const float* __restrict__ v,
                                                 int K, const float* __restrict__ amax_w_slot,
                                                 signed char* __restrict__ wq) {
    const int n = blockIdx.y;
    const int k = blockIdx.x * 256 + threadIdx.x;
    const float sw = amax_w_slot[0] / 31.0f + 1e-8f;
    const float inv = g[n] / sqrtf(v[n] + 1e-5f);
    float q = rintf(w[(size_t)n * K + k] * inv / sw);
    q = fminf(fmaxf(q, -32.0f), 31.0f);
    wq[(size_t)n * K + k] = (signed char)q;
}

__global__ __launch_bounds__(256) void k_absmax(const float* __restrict__ x, size_t n,
                                                float* __restrict__ slot) {
    __shared__ float red[256];
    float mx = 0.0f;
    const size_t stride = (size_t)gridDim.x * 256;
    for (size_t i = (size_t)blockIdx.x * 256 + threadIdx.x; i < n; i += stride)
        mx = fmaxf(mx, fabsf(x[i]));
    red[threadIdx.x] = mx;
    __syncthreads();
    for (int s = 128; s > 0; s >>= 1) {
        if (threadIdx.x < s) red[threadIdx.x] = fmaxf(red[threadIdx.x], red[threadIdx.x + s]);
        __syncthreads();
    }
    if (threadIdx.x == 0)
        atomicMax((unsigned int*)slot, __float_as_uint(red[0]));
}

__global__ __launch_bounds__(256) void k_quant_act(const float* __restrict__ x, size_t n,
                                                   const float* __restrict__ slot, int relu,
                                                   signed char* __restrict__ out) {
    const float s = slot[0] / 31.0f + 1e-8f;
    const float invs = 1.0f / s;
    const size_t stride = (size_t)gridDim.x * 256;
    for (size_t i = (size_t)blockIdx.x * 256 + threadIdx.x; i < n; i += stride) {
        float q = rintf(x[i] * invs);
        q = fminf(fmaxf(q, -32.0f), 31.0f);
        if (relu) q = fmaxf(q, 0.0f);
        out[i] = (signed char)q;
    }
}

__global__ __launch_bounds__(256) void k_gemm_i8(const signed char* __restrict__ A,
                                                 const signed char* __restrict__ Wq,
                                                 const float* __restrict__ bf,
                                                 float* __restrict__ out,
                                                 int K, int N,
                                                 const float* __restrict__ sAct,
                                                 const float* __restrict__ sW,
                                                 float* __restrict__ amaxOut) {
    __shared__ signed char Al[2][TM * LDT];
    __shared__ signed char Wl[2][TN * LDT];
    __shared__ float red[256];

    const int tid  = threadIdx.x;
    const int lane = tid & 31;
    const int wave = tid >> 5;
    const int h    = lane >> 4;
    const int l15  = lane & 15;
    const int wM   = (wave >> 2) * 64;
    const int wN   = (wave & 3) * 32;
    const size_t rowBase = (size_t)blockIdx.x * TM;
    const size_t colBase = (size_t)blockIdx.y * TN;

    const int r0 = tid >> 2;
    const int c0 = (tid & 3) << 4;
    const signed char* gA0 = A  + (rowBase + r0) * K + c0;
    const signed char* gA1 = gA0 + (size_t)64 * K;
    const signed char* gW0 = Wq + (colBase + r0) * K + c0;
    const signed char* gW1 = gW0 + (size_t)64 * K;
    const unsigned lAb[2] = {(unsigned)(size_t)&Al[0][r0 * LDT + c0],
                             (unsigned)(size_t)&Al[1][r0 * LDT + c0]};
    const unsigned lWb[2] = {(unsigned)(size_t)&Wl[0][r0 * LDT + c0],
                             (unsigned)(size_t)&Wl[1][r0 * LDT + c0]};

    v8i acc[4][2];
#pragma unroll
    for (int i = 0; i < 4; ++i)
#pragma unroll
        for (int j = 0; j < 2; ++j)
#pragma unroll
            for (int e = 0; e < 8; ++e) acc[i][j][e] = 0;

    async_ld16(lAb[0],            gA0);
    async_ld16(lAb[0] + 64 * LDT, gA1);
    async_ld16(lWb[0],            gW0);
    async_ld16(lWb[0] + 64 * LDT, gW1);
    wait_async0();
    __syncthreads();

    const int kSteps = K >> 6;
    for (int ks = 0; ks < kSteps; ++ks) {
        const int cur = ks & 1;
        const int nxt = cur ^ 1;
        const bool hasNext = (ks + 1 < kSteps);

        if (hasNext) {
            const int ko = (ks + 1) << 6;
            async_ld16(lAb[nxt],            gA0 + ko);
            async_ld16(lAb[nxt] + 64 * LDT, gA1 + ko);
            async_ld16(lWb[nxt],            gW0 + ko);
            async_ld16(lWb[nxt] + 64 * LDT, gW1 + ko);
            if (ks + 2 < kSteps) {
                __builtin_prefetch(gA0 + ko + 64, 0, 3);
                __builtin_prefetch(gW0 + ko + 64, 0, 3);
            }
        }

        v8i aF[4];
#pragma unroll
        for (int mt = 0; mt < 4; ++mt) {
            const signed char* ab = &Al[cur][(wM + mt * 16 + l15) * LDT + h * 16];
#pragma unroll
            for (int p = 0; p < 4; ++p) {
                unsigned long long d =
                    *(const unsigned long long*)(ab + ((p >> 1) * 32 + (p & 1) * 8));
                aF[mt][2 * p]     = (int)(unsigned int)(d & 0xffffffffull);
                aF[mt][2 * p + 1] = (int)(unsigned int)(d >> 32);
            }
        }
        v8i bF[2];
#pragma unroll
        for (int nt = 0; nt < 2; ++nt) {
            const signed char* bb = &Wl[cur][(wN + nt * 16 + l15) * LDT + h * 16];
            int4 lo = *(const int4*)(bb);
            int4 hi = *(const int4*)(bb + 32);
            bF[nt][0] = lo.x; bF[nt][1] = lo.y; bF[nt][2] = lo.z; bF[nt][3] = lo.w;
            bF[nt][4] = hi.x; bF[nt][5] = hi.y; bF[nt][6] = hi.z; bF[nt][7] = hi.w;
        }
#pragma unroll
        for (int mt = 0; mt < 4; ++mt)
#pragma unroll
            for (int nt = 0; nt < 2; ++nt)
                acc[mt][nt] = __builtin_amdgcn_wmma_i32_16x16x64_iu8(
                    true, aF[mt], true, bF[nt], acc[mt][nt], false, false);

        wait_async0();
        __syncthreads();
    }

    const float sa = sAct[0] / 31.0f + 1e-8f;
    const float sw = sW[0] / 31.0f + 1e-8f;
    const float bs = sa * sw;
    float lmax = 0.0f;
#pragma unroll
    for (int mt = 0; mt < 4; ++mt) {
#pragma unroll
        for (int nt = 0; nt < 2; ++nt) {
            const size_t col = colBase + wN + nt * 16 + l15;
            const float bqv = fminf(fmaxf(rintf(bf[col] / bs), -32.0f), 31.0f) * bs;
#pragma unroll
            for (int r = 0; r < 8; ++r) {
                const size_t row = rowBase + wM + mt * 16 + r + h * 8;
                const float o = (float)acc[mt][nt][r] * bs + bqv;
                out[row * N + col] = o;
                lmax = fmaxf(lmax, fabsf(o));
            }
        }
    }
    red[tid] = lmax;
    __syncthreads();
    for (int s = 128; s > 0; s >>= 1) {
        if (tid < s) red[tid] = fmaxf(red[tid], red[tid + s]);
        __syncthreads();
    }
    if (tid == 0)
        atomicMax((unsigned int*)amaxOut, __float_as_uint(red[0]));
}

__global__ __launch_bounds__(256) void k_softmax(const float* __restrict__ x,
                                                 const float* __restrict__ slot,
                                                 float* __restrict__ out, int N) {
    __shared__ float red[256];
    const int row = blockIdx.x;
    const float* xr = x + (size_t)row * N;
    const float s = slot[0] / 31.0f + 1e-8f;
    float vals[4];
    float mx = -3.4e38f;
#pragma unroll
    for (int i = 0; i < 4; ++i) {
        const int c = threadIdx.x + i * 256;
        float q = fminf(fmaxf(rintf(xr[c] / s), -32.0f), 31.0f) * s;
        vals[i] = q;
        mx = fmaxf(mx, q);
    }
    red[threadIdx.x] = mx;
    __syncthreads();
    for (int t = 128; t > 0; t >>= 1) {
        if (threadIdx.x < t) red[threadIdx.x] = fmaxf(red[threadIdx.x], red[threadIdx.x + t]);
        __syncthreads();
    }
    const float m = red[0];
    __syncthreads();
    float sum = 0.0f;
#pragma unroll
    for (int i = 0; i < 4; ++i) {
        vals[i] = expf(vals[i] - m);
        sum += vals[i];
    }
    red[threadIdx.x] = sum;
    __syncthreads();
    for (int t = 128; t > 0; t >>= 1) {
        if (threadIdx.x < t) red[threadIdx.x] += red[threadIdx.x + t];
        __syncthreads();
    }
    const float invs = 1.0f / red[0];
#pragma unroll
    for (int i = 0; i < 4; ++i)
        out[(size_t)row * N + threadIdx.x + i * 256] = vals[i] * invs;
}

extern "C" void kernel_launch(void* const* d_in, const int* in_sizes, int n_in,
                              void* d_out, int out_size, void* d_ws, size_t ws_size,
                              hipStream_t stream) {
    (void)in_sizes; (void)n_in; (void)out_size; (void)ws_size;
    const float* x = (const float*)d_in[0];
    const float *W[4], *Bv[4], *G[4], *Be[4], *Mn[4], *Vr[4];
    for (int l = 0; l < 4; ++l) {
        W[l]  = (const float*)d_in[1 + 6 * l];
        Bv[l] = (const float*)d_in[2 + 6 * l];
        G[l]  = (const float*)d_in[3 + 6 * l];
        Be[l] = (const float*)d_in[4 + 6 * l];
        Mn[l] = (const float*)d_in[5 + 6 * l];
        Vr[l] = (const float*)d_in[6 + 6 * l];
    }
    const int dims[5] = {1024, 4096, 4096, 4096, 1024};
    const int Bb = 8192;

    char* ws = (char*)d_ws;
    float* amax_act = (float*)(ws);
    float* amax_w   = (float*)(ws + 32);
    float* bfp      = (float*)(ws + 256);
    size_t off = 131072;
    signed char* wq[4];
    for (int l = 0; l < 4; ++l) {
        wq[l] = (signed char*)(ws + off);
        off += (size_t)dims[l + 1] * dims[l];
    }
    off = (off + 255) & ~(size_t)255;
    signed char* act8 = (signed char*)(ws + off);
    off += (size_t)Bb * 4096;
    off = (off + 255) & ~(size_t)255;
    float* actf = (float*)(ws + off);

    k_init<<<1, 64, 0, stream>>>((float*)ws);
    for (int l = 0; l < 4; ++l)
        k_prep<<<dims[l + 1], 256, 0, stream>>>(W[l], Bv[l], G[l], Be[l], Mn[l], Vr[l],
                                                dims[l], bfp + l * 4096, amax_w + l);
    for (int l = 0; l < 4; ++l)
        k_quant_w<<<dim3(dims[l] / 256, dims[l + 1]), 256, 0, stream>>>(
            W[l], G[l], Vr[l], dims[l], amax_w + l, wq[l]);
    k_absmax<<<512, 256, 0, stream>>>(x, (size_t)Bb * 1024, amax_act + 0);
    k_quant_act<<<1024, 256, 0, stream>>>(x, (size_t)Bb * 1024, amax_act + 0, 0, act8);
    k_gemm_i8<<<dim3(Bb / TM, 4096 / TN), 256, 0, stream>>>(
        act8, wq[0], bfp + 0, actf, 1024, 4096, amax_act + 0, amax_w + 0, amax_act + 1);
    k_quant_act<<<2048, 256, 0, stream>>>(actf, (size_t)Bb * 4096, amax_act + 1, 1, act8);
    k_gemm_i8<<<dim3(Bb / TM, 4096 / TN), 256, 0, stream>>>(
        act8, wq[1], bfp + 4096, actf, 4096, 4096, amax_act + 1, amax_w + 1, amax_act + 2);
    k_quant_act<<<2048, 256, 0, stream>>>(actf, (size_t)Bb * 4096, amax_act + 2, 1, act8);
    k_gemm_i8<<<dim3(Bb / TM, 4096 / TN), 256, 0, stream>>>(
        act8, wq[2], bfp + 8192, actf, 4096, 4096, amax_act + 2, amax_w + 2, amax_act + 3);
    k_quant_act<<<2048, 256, 0, stream>>>(actf, (size_t)Bb * 4096, amax_act + 3, 1, act8);
    k_gemm_i8<<<dim3(Bb / TM, 1024 / TN), 256, 0, stream>>>(
        act8, wq[3], bfp + 12288, actf, 4096, 1024, amax_act + 3, amax_w + 3, amax_act + 4);
    k_softmax<<<Bb, 256, 0, stream>>>(actf, amax_act + 4, (float*)d_out, 1024);
}
